// MPNN_11647951306932
// MI455X (gfx1250) — matrix operand order repaired and checked by reading, not compiled, not run on hardware
//
#include <hip/hip_runtime.h>

typedef _Float16 f16;
typedef __attribute__((ext_vector_type(16))) f16   v16h;
typedef __attribute__((ext_vector_type(8)))  f16   v8h;
typedef __attribute__((ext_vector_type(8)))  float v8f;
typedef __attribute__((ext_vector_type(4)))  unsigned int u32x4;
typedef __attribute__((ext_vector_type(8)))  int i32x8;
typedef __attribute__((ext_vector_type(4)))  int i32x4;

#define NB 4
#define NN 512
#define NF 64

// ---------------------------------------------------------------------------
// WMMA helpers (CDNA5 wave32).  A: 16x32 f16, rows striped across VGPRs within
// a lane (lane m holds row m&15; elements 0..7 = K kk+kh+0..7, 8..15 = +16).
// B: 32x16 f16; lane n holds col n&15, same K order as A (kk+kg+0..7, +16).
// C/D: f32, lane l holds (M=(l>>4)*8+v, N=l&15) in 8 VGPRs.
// ---------------------------------------------------------------------------
static __device__ __forceinline__ v8f wmma_f16(v16h a, v16h b, v8f c) {
  return __builtin_amdgcn_wmma_f32_16x16x32_f16(false, a, false, b, (short)0, c,
                                                false, false);
}

static __device__ __forceinline__ v16h ldA16(const f16* p) {
  v8h lo = *(const v8h*)p;          // K = kk+kh .. +7
  v8h hi = *(const v8h*)(p + 16);   // K = kk+16+kh .. +7
  v16h r;
#pragma unroll
  for (int j = 0; j < 8; ++j) { r[j] = lo[j]; r[8 + j] = hi[j]; }
  return r;
}
static __device__ __forceinline__ v16h ldB16(const f16* p) {
  v8h lo = *(const v8h*)p;
  v8h hi = *(const v8h*)(p + 16);
  v16h r;
#pragma unroll
  for (int j = 0; j < 8; ++j) { r[j] = lo[j]; r[8 + j] = hi[j]; }
  return r;
}

// ---------------------------------------------------------------------------
// K1: per-node nonzero counts (row counts for g, column counts for p) and
// reciprocal of nf1 = max(count,1).
// ---------------------------------------------------------------------------
__global__ __launch_bounds__(256) void k_norms(const float* __restrict__ obs,
                                               float* __restrict__ normG,
                                               float* __restrict__ normP,
                                               float* __restrict__ rnfG,
                                               float* __restrict__ rnfP) {
  int bx = blockIdx.x;
  int side = bx >> 3, b = (bx >> 1) & 3;
  int node = (bx & 1) * 256 + threadIdx.x;
  float cnt = 0.f;
  if (side == 0) {
    const float* p = obs + ((size_t)b * NN + node) * NN;
    for (int c = 0; c < NN; ++c) cnt += (p[c] != 0.f) ? 1.f : 0.f;
    normG[b * NN + node] = cnt;
    rnfG[b * NN + node] = 1.f / fmaxf(cnt, 1.f);
  } else {
    const float* p = obs + (size_t)b * NN * NN + node;
    for (int r = 0; r < NN; ++r) cnt += (p[(size_t)r * NN] != 0.f) ? 1.f : 0.f;
    normP[b * NN + node] = cnt;
    rnfP[b * NN + node] = 1.f / fmaxf(cnt, 1.f);
  }
}

// ---------------------------------------------------------------------------
// K2: global maxes of nf1 (per side), folded edge-embedding vectors
// u = relu(W_ee) @ W_ef[0:63,:], and f16-transposed msg/update weights
// WT[l][n][k] for WMMA B operands.
// scal layout: [0]=1/maxG  [1]=1/maxP  [8..71]=u_g  [72..135]=u_p
// ---------------------------------------------------------------------------
__global__ __launch_bounds__(256) void k_scal(
    const float* __restrict__ normG, const float* __restrict__ normP,
    const float* __restrict__ Wee_g, const float* __restrict__ Wef_g,
    const float* __restrict__ Wee_p, const float* __restrict__ Wef_p,
    const float* __restrict__ Wmsg_g, const float* __restrict__ Wupd_g,
    const float* __restrict__ Wmsg_p, const float* __restrict__ Wupd_p,
    float* __restrict__ scal,
    f16* __restrict__ WmTg, f16* __restrict__ WuTg,
    f16* __restrict__ WmTp, f16* __restrict__ WuTp) {
  __shared__ float red[256];
  int tid = threadIdx.x;

  float mg = 1.f;
  for (int i = tid; i < NB * NN; i += 256) mg = fmaxf(mg, normG[i]);
  red[tid] = mg;
  __syncthreads();
  for (int s = 128; s > 0; s >>= 1) {
    if (tid < s) red[tid] = fmaxf(red[tid], red[tid + s]);
    __syncthreads();
  }
  if (tid == 0) scal[0] = 1.f / red[0];
  __syncthreads();

  float mp = 1.f;
  for (int i = tid; i < NB * NN; i += 256) mp = fmaxf(mp, normP[i]);
  red[tid] = mp;
  __syncthreads();
  for (int s = 128; s > 0; s >>= 1) {
    if (tid < s) red[tid] = fmaxf(red[tid], red[tid + s]);
    __syncthreads();
  }
  if (tid == 0) scal[1] = 1.f / red[0];

  if (tid < 64) {
    float a = 0.f;
    for (int k = 0; k < 63; ++k) a += fmaxf(Wee_g[k], 0.f) * Wef_g[k * 64 + tid];
    scal[8 + tid] = a;
  } else if (tid < 128) {
    int j = tid - 64;
    float a = 0.f;
    for (int k = 0; k < 63; ++k) a += fmaxf(Wee_p[k], 0.f) * Wef_p[k * 64 + j];
    scal[72 + j] = a;
  }

  for (int idx = tid; idx < 3 * 128 * 64; idx += 256) {
    int l = idx >> 13, rem = idx & 8191, k = rem >> 6, n = rem & 63;
    int o = (l * 64 + n) * 128 + k;     // WT[l][n][k] = W[l][k][n]
    WmTg[o] = (f16)Wmsg_g[idx];
    WuTg[o] = (f16)Wupd_g[idx];
    WmTp[o] = (f16)Wmsg_p[idx];
    WuTp[o] = (f16)Wupd_p[idx];
  }
}

// ---------------------------------------------------------------------------
// K3: adj -> f16 (row-major) and adj^T -> f16 (for the p-side A operand).
// Values {0,1,2} are exact in f16.
// ---------------------------------------------------------------------------
__global__ __launch_bounds__(256) void k_adj(const float* __restrict__ obs,
                                             f16* __restrict__ adj16,
                                             f16* __restrict__ adjT16) {
  int bx = blockIdx.x;
  int b = bx >> 9, r = bx & 511;
  int c = threadIdx.x * 2;
  const float* p = obs + ((size_t)b * NN + r) * NN + c;
  float x0 = p[0], x1 = p[1];
  size_t o = ((size_t)b * NN + r) * NN + c;
  adj16[o] = (f16)x0;
  adj16[o + 1] = (f16)x1;
  adjT16[((size_t)b * NN + c) * NN + r] = (f16)x0;
  adjT16[((size_t)b * NN + c + 1) * NN + r] = (f16)x1;
}

// ---------------------------------------------------------------------------
// K4: initial node embeddings h and (collapsed) edge features e, both sides.
// h stored row-major [side][B][512][64] and feature-major [side][B][64][512].
// ---------------------------------------------------------------------------
__global__ __launch_bounds__(256) void k_init(
    const float* __restrict__ obs,
    const float* __restrict__ normG, const float* __restrict__ normP,
    const float* __restrict__ rnfG, const float* __restrict__ rnfP,
    const float* __restrict__ scal,
    const float* __restrict__ Winit_g, const float* __restrict__ Winit_p,
    const float* __restrict__ Wef_g, const float* __restrict__ Wef_p,
    f16* __restrict__ e16, f16* __restrict__ h_rm, f16* __restrict__ hT) {
  int bx = blockIdx.x;
  int side = bx >> 3, b = (bx >> 1) & 3;
  int node = (bx & 1) * 256 + threadIdx.x;
  float s = 0.f, nrm, rnf, minv;
  const float *u, *Winit, *Wef;
  if (side == 0) {
    const float* p = obs + ((size_t)b * NN + node) * NN;
    for (int c = 0; c < NN; ++c)
      if (p[c] != 0.f) s += normP[b * NN + c];
    nrm = normG[b * NN + node]; rnf = rnfG[b * NN + node];
    minv = scal[0]; u = scal + 8; Winit = Winit_g; Wef = Wef_g;
  } else {
    const float* p = obs + (size_t)b * NN * NN + node;
    for (int r = 0; r < NN; ++r)
      if (p[(size_t)r * NN] != 0.f) s += normG[b * NN + r];
    nrm = normP[b * NN + node]; rnf = rnfP[b * NN + node];
    minv = scal[1]; u = scal + 72; Winit = Winit_p; Wef = Wef_p;
  }
  s *= (1.f / 512.f) * rnf;            // s = (sum mask*nf_opp)/nf1
  float nf = nrm * (1.f / 512.f);
  float t = nrm * minv;                // norm / max(nf1)
  size_t orow = ((size_t)(side * NB + b) * NN + node) * NF;
  size_t ocol = (size_t)(side * NB + b) * NF;
  for (int j = 0; j < NF; ++j) {
    float hv = fmaxf(nf * Winit[j], 0.f);
    float ev = fmaxf(s * u[j] + t * Wef[63 * 64 + j], 0.f);
    h_rm[orow + j] = (f16)hv;
    hT[(ocol + j) * NN + node] = (f16)hv;
    e16[orow + j] = (f16)ev;
  }
}

// ---------------------------------------------------------------------------
// K5: one MPNN layer, both sides.  Grid: B(4) x stripes(4) x sides(2) = 32
// blocks of 256 threads (8 wave32).  Each wave owns 16 output rows:
//   step1: agg = (adj @ h_opp) * rnf          (16x512x64, 64 WMMA)
//   step2: m   = relu([agg|e] @ W_msg)        (16 WMMA)
//   step3: h'  = relu([h|m]   @ W_upd)        (16 WMMA)
// All 8 waves consume the SAME h_oppT (64KB) and weight slices (2x16KB), so
// they are staged into LDS once per block: h_oppT via the Tensor Data Mover
// (TDM, TENSORcnt) issued by wave 0 and overlapped with the VALU weight copy.
// ---------------------------------------------------------------------------
__global__ __launch_bounds__(256) void k_layer(
    const f16* __restrict__ adj16, const f16* __restrict__ adjT16,
    const f16* __restrict__ hcur_rm, const f16* __restrict__ hcurT,
    f16* __restrict__ hnxt_rm, f16* __restrict__ hnxtT,
    const f16* __restrict__ e16,
    const f16* __restrict__ WmTg, const f16* __restrict__ WmTp,
    const f16* __restrict__ WuTg, const f16* __restrict__ WuTp,
    const float* __restrict__ rnfG, const float* __restrict__ rnfP) {
  __shared__ f16 shH[NF * NN];        // staged h_oppT, 64KB
  __shared__ f16 shWm[64 * 128];      // staged W_msg^T slice, 16KB
  __shared__ f16 shWu[64 * 128];      // staged W_upd^T slice, 16KB
  __shared__ f16 smem[8 * 16 * NF];   // per-wave C->A relayout tiles, 16KB

  int tid = threadIdx.x, wave = tid >> 5, lane = tid & 31;
  int bx = blockIdx.x;
  int side = bx & 1, stripe = (bx >> 1) & 3, b = bx >> 3;
  int row0 = stripe * 128 + wave * 16;

  const f16* A = (side ? adjT16 : adj16) + ((size_t)b * NN + row0) * NN;
  const f16* hoppT = hcurT + (size_t)((1 - side) * NB + b) * NF * NN;
  const f16* hrow = hcur_rm + ((size_t)(side * NB + b) * NN + row0) * NF;
  const f16* erow = e16 + ((size_t)(side * NB + b) * NN + row0) * NF;
  const f16* WmT = side ? WmTp : WmTg;
  const f16* WuT = side ? WuTp : WuTg;
  const float* rnf = (side ? rnfP : rnfG) + b * NN + row0;

  // ---- stage h_oppT (64KB) into LDS via TDM; overlap with weight copy ----
#if __has_builtin(__builtin_amdgcn_tensor_load_to_lds)
  if (wave == 0) {
    // 1-D copy descriptor (D#): 8192 x 8B elements, tile == tensor.
    unsigned lds_off = (unsigned)(uintptr_t)(const void*)&shH[0];
    unsigned long long ga = (unsigned long long)(uintptr_t)hoppT;
    const unsigned NQ = (NF * NN * 2) / 8;  // 8192 qwords
    u32x4 g0 = {1u,                                   // count=1, user mode
                lds_off,                              // lds_addr (bytes)
                (unsigned)ga,                         // global_addr[31:0]
                (unsigned)((ga >> 32) & 0x01FFFFFFu)  // global_addr[56:32]
                    | (2u << 30)};                    // type = 2 ("image")
    i32x8 g1 = {(int)(3u << 16),              // data_size=3 (8B), mask=0
                (int)((NQ & 0xFFFFu) << 16),  // tensor_dim0[15:0]
                (int)((NQ >> 16) | (1u << 16)),  // dim0[31:16] | tensor_dim1=1
                (int)(NQ << 16),              // tile_dim0 = NQ
                1,                            // tile_dim1 = 1
                (int)NQ,                      // tensor_dim0_stride[31:0]
                0, 0};
    i32x4 gz = {0, 0, 0, 0};
#if defined(__clang_major__) && (__clang_major__ >= 23)
    i32x8 gz8 = {0, 0, 0, 0, 0, 0, 0, 0};
    __builtin_amdgcn_tensor_load_to_lds(g0, g1, gz, gz, gz8, 0);
#else
    __builtin_amdgcn_tensor_load_to_lds(g0, g1, gz, gz, 0);
#endif
  }
#endif

  // weight slices -> LDS (all 256 threads; overlaps the TDM transfer)
  {
    const v8h* sm = (const v8h*)WmT;
    const v8h* su = (const v8h*)WuT;
    v8h* dm = (v8h*)shWm;
    v8h* du = (v8h*)shWu;
#pragma unroll
    for (int i = 0; i < (64 * 128) / (8 * 256); ++i) {
      dm[tid + i * 256] = sm[tid + i * 256];
      du[tid + i * 256] = su[tid + i * 256];
    }
  }

#if __has_builtin(__builtin_amdgcn_tensor_load_to_lds)
#if __has_builtin(__builtin_amdgcn_s_wait_tensorcnt)
  if (wave == 0) __builtin_amdgcn_s_wait_tensorcnt((short)0);
#endif
#else
  {  // fallback staging of h_oppT when no TDM builtin is available
    const v8h* src = (const v8h*)hoppT;
    v8h* dst = (v8h*)shH;
    for (int i = tid; i < (NF * NN) / 8; i += 256) dst[i] = src[i];
  }
#endif
  __syncthreads();

  int m = lane & 15;
  int kh = (lane >> 4) << 3;   // A operand K sub-offset
  int kg = (lane >> 4) << 3;   // B operand K sub-offset
  int n = lane & 15;
  int mb = (lane >> 4) << 3;   // C layout row base

  __builtin_prefetch(hrow, 0, 1);   // global_prefetch_b8: pull step2/3 operands
  __builtin_prefetch(erow, 0, 1);

  // ---- step 1: agg = adj @ h_opp ----
  v8f acc[4] = {};
  for (int kk = 0; kk < NN; kk += 32) {
    v16h av = ldA16(A + (size_t)m * NN + kk + kh);
#pragma unroll
    for (int nt = 0; nt < 4; ++nt)
      acc[nt] = wmma_f16(av, ldB16(shH + (nt * 16 + n) * NN + kk + kg), acc[nt]);
  }
  f16* tile = smem + wave * 16 * NF;
#pragma unroll
  for (int v = 0; v < 8; ++v) {
    float r = rnf[mb + v];
#pragma unroll
    for (int nt = 0; nt < 4; ++nt)
      tile[(mb + v) * NF + nt * 16 + n] = (f16)(acc[nt][v] * r);
  }
  __syncthreads();

  // ---- step 2: m = relu([agg|e] @ W_msg) ----
  v8f mm[4] = {};
#pragma unroll
  for (int kk = 0; kk < 64; kk += 32) {
    v16h av = ldA16(tile + m * NF + kk + kh);
#pragma unroll
    for (int nt = 0; nt < 4; ++nt)
      mm[nt] = wmma_f16(av, ldB16(shWm + (nt * 16 + n) * 128 + kk + kg), mm[nt]);
  }
#pragma unroll
  for (int kk = 0; kk < 64; kk += 32) {
    v16h av = ldA16(erow + (size_t)m * NF + kk + kh);
#pragma unroll
    for (int nt = 0; nt < 4; ++nt)
      mm[nt] = wmma_f16(av, ldB16(shWm + (nt * 16 + n) * 128 + 64 + kk + kg),
                        mm[nt]);
  }
  __syncthreads();
#pragma unroll
  for (int v = 0; v < 8; ++v)
#pragma unroll
    for (int nt = 0; nt < 4; ++nt)
      tile[(mb + v) * NF + nt * 16 + n] = (f16)fmaxf(mm[nt][v], 0.f);
  __syncthreads();

  // ---- step 3: h' = relu([h|m] @ W_upd) ----
  v8f hh[4] = {};
#pragma unroll
  for (int kk = 0; kk < 64; kk += 32) {
    v16h av = ldA16(hrow + (size_t)m * NF + kk + kh);
#pragma unroll
    for (int nt = 0; nt < 4; ++nt)
      hh[nt] = wmma_f16(av, ldB16(shWu + (nt * 16 + n) * 128 + kk + kg), hh[nt]);
  }
#pragma unroll
  for (int kk = 0; kk < 64; kk += 32) {
    v16h av = ldA16(tile + m * NF + kk + kh);
#pragma unroll
    for (int nt = 0; nt < 4; ++nt)
      hh[nt] = wmma_f16(av, ldB16(shWu + (nt * 16 + n) * 128 + 64 + kk + kg),
                        hh[nt]);
  }

  size_t orow = (size_t)(side * NB + b) * NN + row0;
  size_t ocol = (size_t)(side * NB + b) * NF;
#pragma unroll
  for (int v = 0; v < 8; ++v) {
    int rr = mb + v;
#pragma unroll
    for (int nt = 0; nt < 4; ++nt) {
      f16 x = (f16)fmaxf(hh[nt][v], 0.f);
      hnxt_rm[(orow + rr) * NF + nt * 16 + n] = x;
      hnxtT[(ocol + nt * 16 + n) * NN + row0 + rr] = x;
    }
  }
}

// ---------------------------------------------------------------------------
// K6: pooling + readout.  hp contributes a per-batch constant c0 to every row;
// fixed-order reductions keep replays deterministic.
// ---------------------------------------------------------------------------
__global__ __launch_bounds__(256) void k_pool(
    const f16* __restrict__ hg, const f16* __restrict__ hp,
    const float* __restrict__ Wpg, const float* __restrict__ Wpp,
    const float* __restrict__ Wro, const float* __restrict__ bro,
    float* __restrict__ out) {
  __shared__ float pg[4][64], pp[4][64], term[64];
  __shared__ float c0s;
  int b = blockIdx.x, tid = threadIdx.x;
  int j = tid & 63, g = tid >> 6;
  const f16* hgB = hg + (size_t)b * NN * NF;
  const f16* hpB = hp + (size_t)b * NN * NF;
  float sg = 0.f, sp = 0.f;
  for (int r = g * 128; r < g * 128 + 128; ++r) {
    sg += (float)hgB[(size_t)r * NF + j];
    sp += (float)hpB[(size_t)r * NF + j];
  }
  pg[g][j] = sg;
  pp[g][j] = sp;
  __syncthreads();
  if (tid < 64) {
    float ssg = (pg[0][tid] + pg[1][tid]) + (pg[2][tid] + pg[3][tid]);
    float ssp = (pp[0][tid] + pp[1][tid]) + (pp[2][tid] + pp[3][tid]);
    pg[0][tid] = ssg * (1.f / 512.f);
    pp[0][tid] = ssp * (1.f / 512.f);
  }
  __syncthreads();
  if (tid < 64) {
    float a = 0.f;
    for (int k = 0; k < 64; ++k)
      a += pg[0][k] * Wpg[k * 64 + tid] + pp[0][k] * Wpp[k * 64 + tid];
    term[tid] = fmaxf(a, 0.f) * Wro[tid];
  }
  __syncthreads();
  if (tid == 0) {
    float c = bro[0];
    for (int k = 0; k < 64; ++k) c += term[k];
    c0s = c;
  }
  __syncthreads();
  for (int r = tid; r < NN; r += 256) {
    float a = c0s;
    for (int jj = 0; jj < NF; ++jj)
      a += fmaxf((float)hgB[(size_t)r * NF + jj], 0.f) * Wro[64 + jj];
    out[b * NN + r] = a;
  }
}

// ---------------------------------------------------------------------------
extern "C" void kernel_launch(void* const* d_in, const int* in_sizes, int n_in,
                              void* d_out, int out_size, void* d_ws,
                              size_t ws_size, hipStream_t stream) {
  (void)in_sizes; (void)n_in; (void)out_size; (void)ws_size;
  const float* obs     = (const float*)d_in[0];
  const float* Winit_g = (const float*)d_in[1];
  const float* Winit_p = (const float*)d_in[2];
  const float* Wee_g   = (const float*)d_in[3];
  const float* Wef_g   = (const float*)d_in[4];
  const float* Wee_p   = (const float*)d_in[5];
  const float* Wef_p   = (const float*)d_in[6];
  const float* Wmsg_g  = (const float*)d_in[7];
  const float* Wupd_g  = (const float*)d_in[8];
  const float* Wmsg_p  = (const float*)d_in[9];
  const float* Wupd_p  = (const float*)d_in[10];
  const float* Wpool_g = (const float*)d_in[11];
  const float* Wpool_p = (const float*)d_in[12];
  const float* Wro     = (const float*)d_in[13];
  const float* bro     = (const float*)d_in[14];
  float* out = (float*)d_out;

  char* w = (char*)d_ws;
  size_t off = 0;
  auto take = [&](size_t bytes) {
    char* p = w + off;
    off += (bytes + 255) & ~(size_t)255;
    return p;
  };
  float* normG = (float*)take(NB * NN * 4);
  float* normP = (float*)take(NB * NN * 4);
  float* rnfG  = (float*)take(NB * NN * 4);
  float* rnfP  = (float*)take(NB * NN * 4);
  float* scal  = (float*)take(1024);
  f16* adj16   = (f16*)take((size_t)NB * NN * NN * 2);
  f16* adjT16  = (f16*)take((size_t)NB * NN * NN * 2);
  f16* WmTg    = (f16*)take(3 * 64 * 128 * 2);
  f16* WuTg    = (f16*)take(3 * 64 * 128 * 2);
  f16* WmTp    = (f16*)take(3 * 64 * 128 * 2);
  f16* WuTp    = (f16*)take(3 * 64 * 128 * 2);
  f16* e16     = (f16*)take((size_t)2 * NB * NN * NF * 2);
  f16* h_rm    = (f16*)take((size_t)2 * 2 * NB * NN * NF * 2);  // [ping][side][B][512][64]
  f16* hT      = (f16*)take((size_t)2 * 2 * NB * NN * NF * 2);  // [ping][side][B][64][512]

  k_norms<<<16, 256, 0, stream>>>(obs, normG, normP, rnfG, rnfP);
  k_scal<<<1, 256, 0, stream>>>(normG, normP, Wee_g, Wef_g, Wee_p, Wef_p,
                                Wmsg_g, Wupd_g, Wmsg_p, Wupd_p, scal,
                                WmTg, WuTg, WmTp, WuTp);
  k_adj<<<NB * NN, 256, 0, stream>>>(obs, adj16, adjT16);
  k_init<<<16, 256, 0, stream>>>(obs, normG, normP, rnfG, rnfP, scal,
                                 Winit_g, Winit_p, Wef_g, Wef_p, e16, h_rm, hT);

  const size_t pingStride = (size_t)2 * NB * NN * NF;
  for (int l = 0; l < 3; ++l) {
    f16* hc_rm = h_rm + (size_t)(l & 1) * pingStride;
    f16* hcT   = hT   + (size_t)(l & 1) * pingStride;
    f16* hn_rm = h_rm + (size_t)((l + 1) & 1) * pingStride;
    f16* hnT   = hT   + (size_t)((l + 1) & 1) * pingStride;
    k_layer<<<32, 256, 0, stream>>>(adj16, adjT16, hc_rm, hcT, hn_rm, hnT, e16,
                                    WmTg + (size_t)l * 64 * 128,
                                    WmTp + (size_t)l * 64 * 128,
                                    WuTg + (size_t)l * 64 * 128,
                                    WuTp + (size_t)l * 64 * 128, rnfG, rnfP);
  }
  const f16* hgF = h_rm + pingStride;                          // ping1, side g
  const f16* hpF = h_rm + pingStride + (size_t)NB * NN * NF;   // ping1, side p
  k_pool<<<NB, 256, 0, stream>>>(hgF, hpF, Wpool_g, Wpool_p, Wro, bro, out);
}
